// GaussianAggregator_81346680586519
// MI455X (gfx1250) — compile-verified, matrix operand order checked by reading, not run on hardware
//
#include <hip/hip_runtime.h>
#include <hip/hip_bf16.h>
#include <math.h>

// ---------------- CDNA5 (gfx1250) types -----------------------------------
typedef __attribute__((ext_vector_type(16))) __bf16 v16bf;
typedef __attribute__((ext_vector_type(8)))  float  v8f;

#define DIM      256
#define LN_EPS   1e-5f
#define WAVES_PW 4            // waves per workgroup (128 threads)
#define FRAG_HW  4096         // bf16 elements per wave frag slice (8 ksteps*32 lanes*16)

__device__ __forceinline__ unsigned short f2bf(float f) {
    __bf16 h = (__bf16)f;
    return __builtin_bit_cast(unsigned short, h);
}

// A/B 16-bit fragment layout for V_WMMA_*_16X16X32:
//   lane l<16 holds row/col l, elements 0..7 -> K 0..7, 8..15 -> K 16..23
//   lane l+16 holds row/col l, elements 0..7 -> K 8..15, 8..15 -> K 24..31
// element (r, j) of a 16 x 256 tile -> halfword index in [kstep][lane][pos]
__device__ __forceinline__ int frag_index(int r, int j) {
    int kstep = j >> 5;
    int jj    = j & 31;
    int lane  = r + (((jj >> 3) & 1) << 4);
    int pos   = (jj & 7) + ((jj >> 4) << 3);
    return (kstep * 32 + lane) * 16 + pos;
}
// inverse: k-within-32 for (lane, pos)
__device__ __forceinline__ int frag_k(int lane, int pos) {
    return (pos & 7) + ((pos >> 3) << 4) + ((lane >> 4) << 3);
}

// ---------------- Prep: swizzle w2 / lf_w into bf16 B-fragment layout ------
// wsB : 16 ntiles * 8 ksteps * 32 lanes * 16 bf16  (w2, 256x256)
// wsLF:  2 ntiles * 8 ksteps * 32 lanes * 16 bf16  (lf_w, 256x18 zero-padded to 32)
__global__ void __launch_bounds__(256)
prep_kernel(const float* __restrict__ w2, const float* __restrict__ lf_w,
            unsigned short* __restrict__ wsB, unsigned short* __restrict__ wsLF) {
    int e = blockIdx.x * blockDim.x + threadIdx.x;
    const int NB = 16 * 8 * 32 * 16;   // 65536
    const int NL =  2 * 8 * 32 * 16;   // 8192
    if (e < NB) {
        int pos = e & 15, lane = (e >> 4) & 31, ks = (e >> 9) & 7, nt = e >> 12;
        int col = nt * 16 + (lane & 15);
        int k   = ks * 32 + frag_k(lane, pos);
        wsB[e] = f2bf(w2[k * DIM + col]);
    } else if (e < NB + NL) {
        int e2 = e - NB;
        int pos = e2 & 15, lane = (e2 >> 4) & 31, ks = (e2 >> 9) & 7, nt = e2 >> 12;
        int col = nt * 16 + (lane & 15);
        int k   = ks * 32 + frag_k(lane, pos);
        wsLF[e2] = f2bf((col < 18) ? lf_w[k * 18 + col] : 0.0f);
    }
}

// ---------------- Fused main kernel ----------------------------------------
__global__ void __launch_bounds__(128)
gauss_agg_kernel(const float* __restrict__ mu,   const float* __restrict__ scale,
                 const float* __restrict__ rot,  const float* __restrict__ Q,
                 const float* __restrict__ w1,   const float* __restrict__ b1,
                 const float* __restrict__ gam,  const float* __restrict__ bet,
                 const float* __restrict__ b2,   const float* __restrict__ lf_b,
                 const unsigned short* __restrict__ wsB,
                 const unsigned short* __restrict__ wsLF,
                 float* __restrict__ out, int Nrows, int numTiles) {
    __shared__ unsigned short hfrag[WAVES_PW * FRAG_HW];   // 32 KB
    __shared__ float          lfbuf[WAVES_PW * 16 * 18];   // 4.5 KB

    const int lane   = threadIdx.x & 31;
    const int wave   = threadIdx.x >> 5;
    const int gwave  = blockIdx.x * WAVES_PW + wave;
    const int nwaves = gridDim.x * WAVES_PW;
    unsigned short* hfw  = hfrag + wave * FRAG_HW;
    float*          lfw_ = lfbuf + wave * 16 * 18;

    const size_t OFF_CORNERS = (size_t)Nrows * DIM;
    const size_t OFF_XY      = OFF_CORNERS + (size_t)Nrows * 33;
    const size_t OFF_XZ      = OFF_XY + (size_t)Nrows * 22;
    const size_t OFF_YZ      = OFF_XZ + (size_t)Nrows * 22;

    // hoist per-lane slice of w1 / b1 / gamma / beta (columns j0..j0+7)
    const int j0 = lane * 8;
    float w1r[10][8], b1r[8], gr[8], br[8];
    #pragma unroll
    for (int k = 0; k < 10; ++k)
        #pragma unroll
        for (int jj = 0; jj < 8; ++jj) w1r[k][jj] = w1[k * DIM + j0 + jj];
    #pragma unroll
    for (int jj = 0; jj < 8; ++jj) {
        b1r[jj] = b1[j0 + jj]; gr[jj] = gam[j0 + jj]; br[jj] = bet[j0 + jj];
    }

    const int colLane = lane & 15;
    const int rowHi   = (lane >> 4) << 3;

    for (int tile = gwave; tile < numTiles; tile += nwaves) {
        const int rowBase = tile * 16;
        // speculative prefetch of next tile's Q stream -> global_prefetch_b8
        {
            int nb = rowBase + 16 * nwaves;
            if (nb < Nrows) __builtin_prefetch(Q + (size_t)nb * DIM, 0, 3);
        }

        // ---- Stage A: h = geo@w1 + b1, LayerNorm, ReLU -> LDS (bf16 A-frag)
        for (int r = 0; r < 16; ++r) {
            int gRow = rowBase + r;
            float geo[10];
            if (gRow < Nrows) {
                geo[0] = mu[gRow * 3 + 0]; geo[1] = mu[gRow * 3 + 1]; geo[2] = mu[gRow * 3 + 2];
                geo[3] = scale[gRow * 3 + 0]; geo[4] = scale[gRow * 3 + 1]; geo[5] = scale[gRow * 3 + 2];
                geo[6] = rot[gRow * 4 + 0]; geo[7] = rot[gRow * 4 + 1];
                geo[8] = rot[gRow * 4 + 2]; geo[9] = rot[gRow * 4 + 3];
            } else {
                #pragma unroll
                for (int k = 0; k < 10; ++k) geo[k] = 0.0f;
            }
            float h[8], s1 = 0.0f, s2 = 0.0f;
            #pragma unroll
            for (int jj = 0; jj < 8; ++jj) {
                float acc = b1r[jj];
                #pragma unroll
                for (int k = 0; k < 10; ++k) acc = fmaf(geo[k], w1r[k][jj], acc);
                h[jj] = acc; s1 += acc; s2 += acc * acc;
            }
            #pragma unroll
            for (int m = 16; m >= 1; m >>= 1) {          // wave32 reduction
                s1 += __shfl_xor(s1, m, 32);
                s2 += __shfl_xor(s2, m, 32);
            }
            float mean = s1 * (1.0f / 256.0f);
            float var  = s2 * (1.0f / 256.0f) - mean * mean;
            float inv  = rsqrtf(var + LN_EPS);
            #pragma unroll
            for (int jj = 0; jj < 8; ++jj) {
                float v = (h[jj] - mean) * inv * gr[jj] + br[jj];
                v = v > 0.0f ? v : 0.0f;
                hfw[frag_index(r, j0 + jj)] = f2bf(v);
            }
        }

        // ---- Stage B: E = h @ w2 (WMMA bf16), features = E + b2 + Q -> out
        for (int nt = 0; nt < 16; ++nt) {
            v8f acc = {};
            #pragma unroll
            for (int ks = 0; ks < 8; ++ks) {
                v16bf a = *(const v16bf*)(hfw + (ks * 32 + lane) * 16);
                v16bf b = *(const v16bf*)(wsB + ((nt * 8 + ks) * 32 + lane) * 16);
                acc = __builtin_amdgcn_wmma_f32_16x16x32_bf16(
                          false, a, false, b, (short)0, acc, false, false);
            }
            int col = nt * 16 + colLane;
            float bb = b2[col];
            #pragma unroll
            for (int v = 0; v < 8; ++v) {
                int rm = v + rowHi, gRow = rowBase + rm;
                if (gRow < Nrows) {
                    float f = acc[v] + bb + Q[(size_t)gRow * DIM + col];
                    out[(size_t)gRow * DIM + col] = f;
                }
            }
        }

        // features are now in L1/L2: make the stores visible to our own gather
        asm volatile("s_wait_storecnt 0x0" ::: "memory");

        // ---- Stage D prologue: gather features back into the (dead) h frag
        {
            int r = colLane, gRow = rowBase + r;
            for (int ks = 0; ks < 8; ++ks) {
                #pragma unroll
                for (int pos = 0; pos < 16; ++pos) {
                    int j = ks * 32 + frag_k(lane, pos);
                    float v = (gRow < Nrows) ? out[(size_t)gRow * DIM + j] : 0.0f;
                    hfw[(ks * 32 + lane) * 16 + pos] = f2bf(v);
                }
            }
        }

        // ---- Stage D: lf = (sigmoid(features @ lf_w + lf_b) - 0.5) * half
        for (int nt = 0; nt < 2; ++nt) {
            v8f acc = {};
            #pragma unroll
            for (int ks = 0; ks < 8; ++ks) {
                v16bf a = *(const v16bf*)(hfw  + (ks * 32 + lane) * 16);
                v16bf b = *(const v16bf*)(wsLF + ((nt * 8 + ks) * 32 + lane) * 16);
                acc = __builtin_amdgcn_wmma_f32_16x16x32_bf16(
                          false, a, false, b, (short)0, acc, false, false);
            }
            int m = nt * 16 + colLane;
            if (m < 18) {
                float lb = lf_b[m];
                int  d3 = m % 3;
                #pragma unroll
                for (int v = 0; v < 8; ++v) {
                    int rm = v + rowHi, gRow = rowBase + rm;
                    float z  = acc[v] + lb;
                    float sg = 1.0f / (1.0f + __expf(-z));
                    float hv = (gRow < Nrows) ? scale[gRow * 3 + d3] * 0.5f : 0.0f;
                    lfw_[rm * 18 + m] = (sg - 0.5f) * hv;
                }
            }
        }

        // ---- Stage E: quaternion -> R, 11 corners, normalized refs ---------
        if (lane < 16) {
            int r = lane, gRow = rowBase + r;
            if (gRow < Nrows) {
                float mx = mu[gRow * 3], my = mu[gRow * 3 + 1], mz = mu[gRow * 3 + 2];
                float sx = scale[gRow * 3], sy = scale[gRow * 3 + 1];
                float qw = rot[gRow * 4], qx = rot[gRow * 4 + 1];
                float qy = rot[gRow * 4 + 2], qz = rot[gRow * 4 + 3];
                float nrm = fmaxf(sqrtf(qw * qw + qx * qx + qy * qy + qz * qz), 1e-8f);
                float iw = qw / nrm, ix = qx / nrm, iy = qy / nrm, iz = qz / nrm;
                float R00 = 1.0f - 2.0f * (iy * iy + iz * iz), R01 = 2.0f * (ix * iy - iz * iw),
                      R02 = 2.0f * (ix * iz + iy * iw);
                float R10 = 2.0f * (ix * iy + iz * iw), R11 = 1.0f - 2.0f * (ix * ix + iz * iz),
                      R12 = 2.0f * (iy * iz - ix * iw);
                float R20 = 2.0f * (ix * iz - iy * iw), R21 = 2.0f * (iy * iz + ix * iw),
                      R22 = 1.0f - 2.0f * (ix * ix + iy * iy);
                float hl = sx * 0.5f, hw2 = sy * 0.5f;
                const float rXY = 1.0f / (102.4f + 1e-6f);
                const float rZ  = 1.0f / (8.0f + 1e-6f);
                for (int p = 0; p < 11; ++p) {
                    float px, py, pz;
                    if      (p == 0) { px = 0;   py = 0;    pz = 0; }
                    else if (p == 1) { px = 0;   py = hw2;  pz = 0; }
                    else if (p == 2) { px = 0;   py = -hw2; pz = 0; }
                    else if (p == 3) { px = hl;  py = 0;    pz = 0; }
                    else if (p == 4) { px = -hl; py = 0;    pz = 0; }
                    else {
                        int kq = p - 5;
                        px = lfw_[r * 18 + kq * 3];
                        py = lfw_[r * 18 + kq * 3 + 1];
                        pz = lfw_[r * 18 + kq * 3 + 2];
                    }
                    float cx = R00 * px + R01 * py + R02 * pz + mx;
                    float cy = R10 * px + R11 * py + R12 * pz + my;
                    float cz = R20 * px + R21 * py + R22 * pz + mz;
                    size_t cb = OFF_CORNERS + (size_t)gRow * 33 + p * 3;
                    out[cb] = cx; out[cb + 1] = cy; out[cb + 2] = cz;
                    float nx = fminf(fmaxf((cx + 51.2f) * rXY, 0.0f), 1.0f);
                    float ny = fminf(fmaxf((cy + 51.2f) * rXY, 0.0f), 1.0f);
                    float nz = fminf(fmaxf((cz + 5.0f)  * rZ,  0.0f), 1.0f);
                    size_t rb = (size_t)gRow * 22 + p * 2;
                    out[OFF_XY + rb] = nx; out[OFF_XY + rb + 1] = ny;
                    out[OFF_XZ + rb] = nx; out[OFF_XZ + rb + 1] = nz;
                    out[OFF_YZ + rb] = ny; out[OFF_YZ + rb + 1] = nz;
                }
            }
        }
    }
}

extern "C" void kernel_launch(void* const* d_in, const int* in_sizes, int n_in,
                              void* d_out, int out_size, void* d_ws, size_t ws_size,
                              hipStream_t stream) {
    const float* mu    = (const float*)d_in[0];
    const float* scale = (const float*)d_in[1];
    const float* rot   = (const float*)d_in[2];
    const float* Q     = (const float*)d_in[3];
    const float* w1    = (const float*)d_in[4];
    const float* b1    = (const float*)d_in[5];
    const float* gam   = (const float*)d_in[6];
    const float* bet   = (const float*)d_in[7];
    const float* w2    = (const float*)d_in[8];
    const float* b2    = (const float*)d_in[9];
    const float* lf_w  = (const float*)d_in[10];
    const float* lf_b  = (const float*)d_in[11];
    // d_in[12] (tpv_spatial_shapes) unused by the math

    const int Nrows = in_sizes[0] / 3;
    unsigned short* wsB  = (unsigned short*)d_ws;                  // 128 KB
    unsigned short* wsLF = wsB + 16 * 8 * 32 * 16;                 //  16 KB
    const int prepElems  = 16 * 8 * 32 * 16 + 2 * 8 * 32 * 16;

    hipLaunchKernelGGL(prep_kernel, dim3((prepElems + 255) / 256), dim3(256), 0, stream,
                       w2, lf_w, wsB, wsLF);

    const int numTiles = (Nrows + 15) / 16;
    hipLaunchKernelGGL(gauss_agg_kernel, dim3(1024), dim3(128), 0, stream,
                       mu, scale, rot, Q, w1, b1, gam, bet, b2, lf_b,
                       wsB, wsLF, (float*)d_out, Nrows, numTiles);
}
